// GATNodeClassifier_26731876451142
// MI455X (gfx1250) — compile-verified, matrix operand order checked by reading, not run on hardware
//
#include <hip/hip_runtime.h>
#include <stdint.h>

typedef __attribute__((ext_vector_type(16))) __bf16 v16bf;
typedef __attribute__((ext_vector_type(4)))  __bf16 v4bf;
typedef __attribute__((ext_vector_type(8)))  float  v8f;

#define NEG_SLOPE 0.2f
#define EPS_LN    1e-5f

// ---- monotone float<->uint encoding for atomicMax on floats ----
__device__ __forceinline__ unsigned f2o(float f) {
  unsigned u = __float_as_uint(f);
  return (u & 0x80000000u) ? ~u : (u | 0x80000000u);
}
__device__ __forceinline__ float o2f(unsigned u) {
  return __uint_as_float((u & 0x80000000u) ? (u & 0x7fffffffu) : ~u);
}

__global__ void fill_u32_kernel(uint32_t* __restrict__ p, uint32_t v, int n) {
  int i = blockIdx.x * blockDim.x + threadIdx.x;
  if (i < n) p[i] = v;
}

// ============================================================================
// bf16 WMMA GEMM: C[M,Nc] = A[M,K] @ B[K,Nc], f32 in/out, bf16 compute.
// Block = 256 threads (8 waves). Block tile 64(M) x 64(N). K step 32.
// Wave (wm 0..3, wn 0..1) owns two 16x16 tiles: (wm, wn*32) and (wm, wn*32+16),
// sharing one A fragment (2 WMMAs / wave / k-step).
// LDS rows are k-ascending bf16; a lane's WMMA fragment is exactly two
// contiguous 16-byte chunks -> two ds_load_b128 per fragment.
// Requires: K % 32 == 0, Nc % 64 == 0 (true for all calls here). M guarded.
// ============================================================================
__global__ __launch_bounds__(256)
void gemm_bf16_wmma(const float* __restrict__ A, const float* __restrict__ B,
                    float* __restrict__ C, int M, int K, int Nc) {
  __shared__ __align__(16) __bf16 sA[64][40];   // [m][k], 80B row stride
  __shared__ __align__(16) __bf16 sB[64][40];   // transposed: [n][k]
  const int tid  = threadIdx.x;
  const int lane = tid & 31, wave = tid >> 5;
  const int wm = wave & 3;
  const int wn = wave >> 2;
  const int mBase = blockIdx.x * 64;
  const int nBase = blockIdx.y * 64;

  // staging coordinates
  const int ar = tid >> 3;          // A row 0..31 (+32 on 2nd iter)
  const int ak = (tid & 7) * 4;     // A k offset 0..28
  const int bn = tid & 63;          // B column 0..63
  const int bk = (tid >> 6) * 4;    // B k offset 0,4,8,12 (+16 on 2nd iter)

  const int mrow = lane & 15;
  const int hf   = lane >> 4;

  v8f acc0 = {}, acc1 = {};

  for (int k0 = 0; k0 < K; k0 += 32) {
    // ---- stage A: 64x32 f32 -> bf16, one float4 + one b64 LDS store x2 ----
#pragma unroll
    for (int i = 0; i < 2; ++i) {
      int r   = ar + i * 32;
      int row = mBase + r;
      row = (row < M) ? row : (M - 1);  // clamp: garbage rows never stored
      const float4 v = *(const float4*)(A + (size_t)row * K + k0 + ak);
      v4bf pk;
      pk[0] = (__bf16)v.x; pk[1] = (__bf16)v.y;
      pk[2] = (__bf16)v.z; pk[3] = (__bf16)v.w;
      *(v4bf*)(&sA[r][ak]) = pk;
    }
    // ---- stage B transposed: 4 coalesced strided f32 -> one b64 store x2 ----
#pragma unroll
    for (int i = 0; i < 2; ++i) {
      int kc = bk + i * 16;
      const float* bp = B + (size_t)(k0 + kc) * Nc + nBase + bn;
      float f0 = bp[0];
      float f1 = bp[(size_t)Nc];
      float f2 = bp[(size_t)Nc * 2];
      float f3 = bp[(size_t)Nc * 3];
      v4bf pk;
      pk[0] = (__bf16)f0; pk[1] = (__bf16)f1;
      pk[2] = (__bf16)f2; pk[3] = (__bf16)f3;
      *(v4bf*)(&sB[bn][kc]) = pk;
    }
    __syncthreads();

    // ---- fragments: two ds_load_b128 each, bit-cast to v16bf ----
    union Cvt { uint4 u[2]; v16bf v; };
    Cvt c;
    const __bf16* pa = &sA[wm * 16 + mrow][0];
    c.u[0] = *(const uint4*)(pa + hf * 8);
    c.u[1] = *(const uint4*)(pa + 16 + hf * 8);
    v16bf fa = c.v;

    const __bf16* pb0 = &sB[wn * 32 + mrow][0];
    c.u[0] = *(const uint4*)(pb0 + hf * 8);
    c.u[1] = *(const uint4*)(pb0 + 16 + hf * 8);
    v16bf fb0 = c.v;

    const __bf16* pb1 = &sB[wn * 32 + 16 + mrow][0];
    c.u[0] = *(const uint4*)(pb1 + hf * 8);
    c.u[1] = *(const uint4*)(pb1 + 16 + hf * 8);
    v16bf fb1 = c.v;

    acc0 = __builtin_amdgcn_wmma_f32_16x16x32_bf16(
        false, fa, false, fb0, (short)0, acc0, false, false);
    acc1 = __builtin_amdgcn_wmma_f32_16x16x32_bf16(
        false, fa, false, fb1, (short)0, acc1, false, false);
    __syncthreads();
  }

  // C/D layout: lane col = lane&15; rows rb..rb+7 where rb = (lane>>4)*8
  const int col0 = nBase + wn * 32 + mrow;
  const int rb   = hf * 8;
#pragma unroll
  for (int r = 0; r < 8; ++r) {
    int row = mBase + wm * 16 + rb + r;
    if (row < M) {
      C[(size_t)row * Nc + col0]      = acc0[r];
      C[(size_t)row * Nc + col0 + 16] = acc1[r];
    }
  }
}

// ============================================================================
// Edge pass 1: e = att . leaky_relu(xl[src] + xr[dst]); segment max via
// ordered-uint atomicMax. One wave per (edge, head).
// ============================================================================
__global__ void edge_score(const float* __restrict__ xl, const float* __restrict__ xr,
                           const float* __restrict__ att, const int* __restrict__ ei,
                           int E, int ET, int H, int C,
                           float* __restrict__ ebuf, unsigned* __restrict__ mmax) {
  int g = blockIdx.x * (blockDim.x >> 5) + (threadIdx.x >> 5);
  if (g >= ET * H) return;
  int e = g / H, h = g % H;
  int s, d;
  if (e < E) { s = ei[e]; d = ei[E + e]; } else { s = d = e - E; }
  int lane = threadIdx.x & 31;
  float acc = 0.0f;
  for (int c = lane; c < C; c += 32) {
    float z = xl[(size_t)s * H * C + h * C + c] + xr[(size_t)d * H * C + h * C + c];
    z = (z > 0.0f) ? z : NEG_SLOPE * z;
    acc += z * att[h * C + c];
  }
#pragma unroll
  for (int off = 16; off > 0; off >>= 1) acc += __shfl_xor(acc, off, 32);
  if (lane == 0) {
    ebuf[g] = acc;
    atomicMax(&mmax[d * H + h], f2o(acc));
  }
}

// ============================================================================
// Edge pass 2: p = exp(e - m[dst]); denom[dst] += p (atomic). Thread per (e,h).
// ============================================================================
__global__ void edge_p(float* __restrict__ ebuf, const unsigned* __restrict__ mmax,
                       float* __restrict__ denom, const int* __restrict__ ei,
                       int E, int ET, int H) {
  int g = blockIdx.x * blockDim.x + threadIdx.x;
  if (g >= ET * H) return;
  int e = g / H, h = g % H;
  int d = (e < E) ? ei[E + e] : (e - E);
  float m = o2f(mmax[d * H + h]);
  float p = __expf(ebuf[g] - m);
  ebuf[g] = p;
  atomicAdd(&denom[d * H + h], p);
}

// ============================================================================
// Edge pass 3: agg[dst] += alpha * xl[src]. Block (H*C threads) per edge.
// ============================================================================
__global__ void edge_scatter(const float* __restrict__ ebuf, const float* __restrict__ denom,
                             const float* __restrict__ xl, const int* __restrict__ ei,
                             int E, int H, int C, float* __restrict__ agg) {
  int e = blockIdx.x;
  int t = threadIdx.x;
  int h = t / C;
  int s, d;
  if (e < E) { s = ei[e]; d = ei[E + e]; } else { s = d = e - E; }
  float alpha = ebuf[(size_t)e * H + h] / (denom[d * H + h] + 1e-16f);
  atomicAdd(&agg[(size_t)d * H * C + t], alpha * xl[(size_t)s * H * C + t]);
}

// ============================================================================
// LayerNorm (+bias in, relu out). Block = HC threads, one node per block.
// ============================================================================
__global__ void ln_relu(const float* __restrict__ agg, const float* __restrict__ bias,
                        const float* __restrict__ gamma, const float* __restrict__ beta,
                        float* __restrict__ out, int HC) {
  extern __shared__ float sred[];
  int node = blockIdx.x, t = threadIdx.x;
  float v = agg[(size_t)node * HC + t] + bias[t];
  sred[t] = v;
  __syncthreads();
  for (int off = HC >> 1; off > 0; off >>= 1) {
    if (t < off) sred[t] += sred[t + off];
    __syncthreads();
  }
  float mu = sred[0] / (float)HC;
  __syncthreads();
  float dv = v - mu;
  sred[t] = dv * dv;
  __syncthreads();
  for (int off = HC >> 1; off > 0; off >>= 1) {
    if (t < off) sred[t] += sred[t + off];
    __syncthreads();
  }
  float var = sred[0] / (float)HC;
  float y = dv * rsqrtf(var + EPS_LN) * gamma[t] + beta[t];
  out[(size_t)node * HC + t] = fmaxf(y, 0.0f);
}

__global__ void bias_relu(float* __restrict__ x, const float* __restrict__ b,
                          int C, int total) {
  int i = blockIdx.x * blockDim.x + threadIdx.x;
  if (i < total) x[i] = fmaxf(x[i] + b[i % C], 0.0f);
}

// Final classifier head: out[node] = dot(h1row, cW2) + cb2. Block = 64.
__global__ void final_dot(const float* __restrict__ h, const float* __restrict__ w,
                          const float* __restrict__ b, float* __restrict__ out, int C) {
  __shared__ float sred[64];
  int node = blockIdx.x, t = threadIdx.x;
  sred[t] = h[(size_t)node * C + t] * w[t];
  __syncthreads();
  for (int off = 32; off > 0; off >>= 1) {
    if (t < off) sred[t] += sred[t + off];
    __syncthreads();
  }
  if (t == 0) out[node] = sred[0] + b[0];
}

// ============================================================================
extern "C" void kernel_launch(void* const* d_in, const int* in_sizes, int n_in,
                              void* d_out, int out_size, void* d_ws, size_t ws_size,
                              hipStream_t stream) {
  const float* x    = (const float*)d_in[0];
  const int*   ei   = (const int*)d_in[1];
  const float* Wl0  = (const float*)d_in[2];
  const float* Wr0  = (const float*)d_in[3];
  const float* att0 = (const float*)d_in[4];
  const float* b0   = (const float*)d_in[5];
  const float* g0   = (const float*)d_in[6];
  const float* be0  = (const float*)d_in[7];
  const float* Wl1  = (const float*)d_in[8];
  const float* Wr1  = (const float*)d_in[9];
  const float* att1 = (const float*)d_in[10];
  const float* b1   = (const float*)d_in[11];
  const float* g1   = (const float*)d_in[12];
  const float* be1  = (const float*)d_in[13];
  const float* cW1  = (const float*)d_in[14];
  const float* cb1  = (const float*)d_in[15];
  const float* cW2  = (const float*)d_in[16];
  const float* cb2  = (const float*)d_in[17];
  float* out = (float*)d_out;

  const int DIN = 1280, HID = 64, HEADS = 4;
  const int N   = in_sizes[0] / DIN;
  const int E   = in_sizes[1] / 2;
  const int ET  = E + N;          // with self loops
  const int HC0 = HEADS * HID;    // 256

  // ---- workspace carve-out (reused across phases) ----
  char* wp = (char*)d_ws;
  auto carve = [&](size_t bytes) -> char* {
    char* p = wp;
    wp += (bytes + 255) & ~(size_t)255;
    return p;
  };
  float*    buf0  = (float*)carve((size_t)N * HC0 * 4);      // xl0 -> h0 -> c1
  float*    buf1  = (float*)carve((size_t)N * HC0 * 4);      // xr0 -> xl1|xr1
  float*    agg   = (float*)carve((size_t)N * HC0 * 4);      // agg0 -> agg1|h1
  float*    ebuf  = (float*)carve((size_t)ET * HEADS * 4);
  unsigned* mmax  = (unsigned*)carve((size_t)N * HEADS * 4);
  float*    denom = (float*)carve((size_t)N * HEADS * 4);

  const unsigned ENC_NEG_INF = 0x007FFFFFu;  // f2o(-inf)
  dim3 blk(256);

  // ================= Layer 0 =================
  {
    dim3 grid((N + 63) / 64, HC0 / 64);
    gemm_bf16_wmma<<<grid, blk, 0, stream>>>(x, Wl0, buf0, N, DIN, HC0);  // xl0
    gemm_bf16_wmma<<<grid, blk, 0, stream>>>(x, Wr0, buf1, N, DIN, HC0);  // xr0

    int nm = N * HEADS;
    fill_u32_kernel<<<(nm + 255) / 256, blk, 0, stream>>>(mmax, ENC_NEG_INF, nm);
    fill_u32_kernel<<<(nm + 255) / 256, blk, 0, stream>>>((uint32_t*)denom, 0u, nm);
    int na = N * HC0;
    fill_u32_kernel<<<(na + 255) / 256, blk, 0, stream>>>((uint32_t*)agg, 0u, na);

    int tot = ET * HEADS;
    edge_score<<<(tot + 7) / 8, blk, 0, stream>>>(buf0, buf1, att0, ei, E, ET,
                                                  HEADS, HID, ebuf, mmax);
    edge_p<<<(tot + 255) / 256, blk, 0, stream>>>(ebuf, mmax, denom, ei, E, ET, HEADS);
    edge_scatter<<<ET, dim3(HC0), 0, stream>>>(ebuf, denom, buf0, ei, E, HEADS, HID, agg);

    // h0 overwrites buf0 (xl0 dead after scatter)
    ln_relu<<<N, HC0, HC0 * 4, stream>>>(agg, b0, g0, be0, buf0, HC0);
  }

  // ================= Layer 1 (heads=1) =================
  float* h1;
  {
    float* xl1 = buf1;
    float* xr1 = buf1 + (size_t)N * HID;
    dim3 grid((N + 63) / 64, HID / 64);
    gemm_bf16_wmma<<<grid, blk, 0, stream>>>(buf0, Wl1, xl1, N, HC0, HID);
    gemm_bf16_wmma<<<grid, blk, 0, stream>>>(buf0, Wr1, xr1, N, HC0, HID);

    fill_u32_kernel<<<(N + 255) / 256, blk, 0, stream>>>(mmax, ENC_NEG_INF, N);
    fill_u32_kernel<<<(N + 255) / 256, blk, 0, stream>>>((uint32_t*)denom, 0u, N);
    int na = N * HID;
    fill_u32_kernel<<<(na + 255) / 256, blk, 0, stream>>>((uint32_t*)agg, 0u, na);

    int tot = ET;
    edge_score<<<(tot + 7) / 8, blk, 0, stream>>>(xl1, xr1, att1, ei, E, ET,
                                                  1, HID, ebuf, mmax);
    edge_p<<<(tot + 255) / 256, blk, 0, stream>>>(ebuf, mmax, denom, ei, E, ET, 1);
    edge_scatter<<<ET, dim3(HID), 0, stream>>>(ebuf, denom, xl1, ei, E, 1, HID, agg);

    h1 = agg + (size_t)N * HID;  // disjoint from agg[0 : N*HID)
    ln_relu<<<N, HID, HID * 4, stream>>>(agg, b1, g1, be1, h1, HID);
  }

  // ================= Classifier MLP =================
  {
    float* c1 = buf0;  // h0 dead after layer-1 GEMMs
    dim3 grid((N + 63) / 64, HID / 64);
    gemm_bf16_wmma<<<grid, blk, 0, stream>>>(h1, cW1, c1, N, HID, HID);
    int tot = N * HID;
    bias_relu<<<(tot + 255) / 256, blk, 0, stream>>>(c1, cb1, HID, tot);
    final_dot<<<N, dim3(HID), 0, stream>>>(c1, cW2, cb2, out, HID);
  }
}
